// InternLM2Attention_36747740184754
// MI455X (gfx1250) — matrix operand order repaired and checked by reading, not compiled, not run on hardware
//
#include <hip/hip_runtime.h>
#include <hip/hip_bf16.h>

// ---------------------------------------------------------------------------
// InternLM2 attention block on gfx1250 (MI455X).
// bf16 WMMA for QKV GEMM, flash attention, O-proj; TDM async V staging.
// B=2, S=2048, HID=4096, NH=32, NKV=8, D=128, G=4.
// ---------------------------------------------------------------------------

typedef __bf16 bf16;
typedef __bf16 v4bf  __attribute__((ext_vector_type(4)));
typedef __bf16 v8bf  __attribute__((ext_vector_type(8)));
typedef __bf16 v16bf __attribute__((ext_vector_type(16)));
typedef float  v8f   __attribute__((ext_vector_type(8)));
typedef unsigned int v4u __attribute__((ext_vector_type(4)));
typedef int    v8i   __attribute__((ext_vector_type(8)));
typedef int    v4i   __attribute__((ext_vector_type(4)));

#define BS     4096      // B*S
#define HID    4096
#define QKVC   6144      // NH*D + 2*NKV*D
#define SEQ    2048
#define NHEAD  32
#define DHEAD  128
#define ATT_SCALE 0.08838834764831845f   // 1/sqrt(128)
#define NEGINF (-1e30f)

// ---- fragment helpers ------------------------------------------------------

__device__ __forceinline__ v16bf frag16(const bf16* p0, const bf16* p1) {
  v8bf lo = *(const v8bf*)p0;
  v8bf hi = *(const v8bf*)p1;
  return __builtin_shufflevector(lo, hi, 0,1,2,3,4,5,6,7,8,9,10,11,12,13,14,15);
}

__device__ __forceinline__ v8f wmma_bf16(v16bf a, v16bf b, v8f c) {
  return __builtin_amdgcn_wmma_f32_16x16x32_bf16(
      false, a, false, b, (short)0, c, false, false);
}

// ---- Tensor Data Mover: 2D bf16 tile (tile_d x tile_rows) global -> LDS ----
// D# per CDNA5 ISA. Tracked by TENSORcnt.

__device__ __forceinline__ void tdm_load_tile_bf16(unsigned lds_addr, const void* gptr,
                                                   int tile_d, int tile_rows,
                                                   int row_stride_elems) {
  unsigned long long ga = (unsigned long long)(uintptr_t)gptr;
  v4u g0;
  g0[0] = 1u;                                                // count=1 (valid), user mode
  g0[1] = lds_addr;                                          // LDS byte address
  g0[2] = (unsigned)(ga & 0xFFFFFFFFu);                      // global_addr[31:0]
  g0[3] = (unsigned)((ga >> 32) & 0x1FFFFFFu) | (2u << 30);  // global_addr[56:32], type=2
  v8i g1;
  g1[0] = (1 << 16);                                         // data_size=1 (2B), wg_mask=0
  g1[1] = (tile_d & 0xFFFF) << 16;                           // tensor_dim0[15:0] @ bits 63:48
  g1[2] = ((tile_d >> 16) & 0xFFFF) |                        // tensor_dim0[31:16]
          ((tile_rows & 0xFFFF) << 16);                      // tensor_dim1[15:0]
  g1[3] = ((tile_rows >> 16) & 0xFFFF) |                     // tensor_dim1[31:16]
          ((tile_d & 0xFFFF) << 16);                         // tile_dim0
  g1[4] = (tile_rows & 0xFFFF);                              // tile_dim1 (tile_dim2=0)
  g1[5] = row_stride_elems;                                  // tensor_dim0_stride[31:0]
  g1[6] = 0;
  g1[7] = 0;
  v4i z4 = {0, 0, 0, 0};                                     // groups 2/3 unused (2D tile)
  v8i z8 = {0, 0, 0, 0, 0, 0, 0, 0};
  __builtin_amdgcn_tensor_load_to_lds(g0, g1, z4, z4, z8, 0);
}

// ---- kernel 1: fp32 -> bf16 elementwise (hidden states) --------------------

__global__ __launch_bounds__(256) void k_cvt_bf16(const float* __restrict__ in,
                                                  bf16* __restrict__ out, int n) {
  int i = (blockIdx.x * 256 + threadIdx.x) * 4;
  if (i < n) {
    float4 v = *(const float4*)(in + i);
    v4bf o = { (bf16)v.x, (bf16)v.y, (bf16)v.z, (bf16)v.w };
    *(v4bf*)(out + i) = o;
  }
}

// ---- kernel 2: fp32 [R,C] -> bf16 [C,R] transpose via LDS tile -------------

__global__ __launch_bounds__(256) void k_transpose_cvt(const float* __restrict__ in,
                                                       bf16* __restrict__ out,
                                                       int R, int C) {
  __shared__ float t[32][33];
  const int tx = threadIdx.x & 31;
  const int ty = threadIdx.x >> 5;
  const int cb = blockIdx.x * 32;
  const int rb = blockIdx.y * 32;
#pragma unroll
  for (int i = 0; i < 4; ++i)
    t[ty + i * 8][tx] = in[(size_t)(rb + ty + i * 8) * C + cb + tx];
  __syncthreads();
#pragma unroll
  for (int i = 0; i < 4; ++i)
    out[(size_t)(cb + ty + i * 8) * R + rb + tx] = (bf16)t[tx][ty + i * 8];
}

// ---- kernel 3: QKV GEMM (bf16 WMMA) + fused neox RoPE ----------------------

__global__ __launch_bounds__(128) void k_qkv_rope(const bf16* __restrict__ X,   // [BS, HID]
                                                  const bf16* __restrict__ WT,  // [QKVC, HID]
                                                  const int*  __restrict__ pos, // [BS]
                                                  bf16* __restrict__ qkv) {     // [BS, QKVC]
  const int wave = threadIdx.x >> 5;
  const int lane = threadIdx.x & 31;
  const int rowTile = blockIdx.x;                 // 0..255
  const int pair = blockIdx.y * 4 + wave;         // 0..191
  int base, head, t;  bool doRope;
  if (pair < 128)      { base = 0;    head = pair >> 2;         t = pair & 3;         doRope = true;  }
  else if (pair < 160) { base = 4096; head = (pair - 128) >> 2; t = (pair - 128) & 3; doRope = true;  }
  else                 { base = 5120; head = (pair - 160) >> 2; t = (pair - 160) & 3; doRope = false; }
  const int c0 = base + head * DHEAD + t * 16;
  const int c1 = c0 + 64;                          // neox partner half
  const int m  = lane & 15;
  const int hi = lane >> 4;

  const bf16* Arow  = X  + (size_t)(rowTile * 16 + m) * HID;
  const bf16* B0row = WT + (size_t)(c0 + m) * HID;
  const bf16* B1row = WT + (size_t)(c1 + m) * HID;

  v8f acc0 = {}; v8f acc1 = {};
  for (int k0 = 0; k0 < HID; k0 += 32) {
    __builtin_prefetch(Arow + k0 + 512, 0, 0);
    __builtin_prefetch(B0row + k0 + 512, 0, 0);
    v16bf a  = frag16(Arow  + k0 + hi * 16, Arow  + k0 + hi * 16 + 8);
    v16bf b0 = frag16(B0row + k0 + hi * 16, B0row + k0 + hi * 16 + 8);
    v16bf b1 = frag16(B1row + k0 + hi * 16, B1row + k0 + hi * 16 + 8);
    acc0 = wmma_bf16(a, b0, acc0);
    acc1 = wmma_bf16(a, b1, acc1);
  }

  const float inv_freq = __expf(-0.14391565298402176f * (float)(t * 16 + m)); // ln(1e4)/64
#pragma unroll
  for (int r = 0; r < 8; ++r) {
    const int mrow = rowTile * 16 + r + 8 * hi;
    float o0 = acc0[r], o1 = acc1[r];
    if (doRope) {
      const float p = (float)pos[mrow];
      const float ang = p * inv_freq;
      const float c = __cosf(ang), s = __sinf(ang);
      const float x1 = o0, x2 = o1;
      o0 = x1 * c - x2 * s;
      o1 = x1 * s + x2 * c;
    }
    qkv[(size_t)mrow * QKVC + c0 + m] = (bf16)o0;
    qkv[(size_t)mrow * QKVC + c1 + m] = (bf16)o1;
  }
}

// ---- kernel 4: causal flash attention, TDM double-buffered V tiles ---------
// grid = (S/16/4, B*NH), block = 128 (4 waves, each an independent q-tile).

__global__ __launch_bounds__(128) void k_attn(const bf16* __restrict__ qkv,  // [BS, QKVC]
                                              bf16* __restrict__ aout) {     // [BS, NH*D]
  __shared__ bf16 VT[4][2][32 * DHEAD];  // per-wave double-buffered V tile [key][d]
  __shared__ bf16 PT[4][16 * 32];        // per-wave P tile [m][key]
  const int wave = threadIdx.x >> 5;
  const int lane = threadIdx.x & 31;
  const int qt  = blockIdx.x * 4 + wave;   // 0..127
  const int bh  = blockIdx.y;              // 0..63
  const int b   = bh >> 5;
  const int h   = bh & 31;
  const int kvh = h >> 2;                  // GQA group
  const size_t rowbase = (size_t)b * SEQ;
  const int m  = lane & 15;
  const int hi = lane >> 4;

  bf16* pt = PT[wave];
  const unsigned vt_lds0 = (unsigned)(uintptr_t)(&VT[wave][0][0]);

  // Q A-fragments over D=128
  const bf16* Qrow = qkv + (rowbase + qt * 16 + m) * QKVC + h * DHEAD;
  v16bf qf[4];
#pragma unroll
  for (int ks = 0; ks < 4; ++ks)
    qf[ks] = frag16(Qrow + ks * 32 + hi * 16, Qrow + ks * 32 + hi * 16 + 8);

  v8f acc[8];
#pragma unroll
  for (int dt = 0; dt < 8; ++dt) acc[dt] = (v8f){};
  float mstat[8], lstat[8];
#pragma unroll
  for (int r = 0; r < 8; ++r) { mstat[r] = NEGINF; lstat[r] = 0.f; }

  const int niter = qt / 2 + 1;            // causal: keys 0 .. qt*16+15
  const bf16* Vbase = qkv + rowbase * QKVC + 5120 + kvh * DHEAD;

  // prime the TDM pipeline with tile 0
  tdm_load_tile_bf16(vt_lds0, Vbase, DHEAD, 32, QKVC);

  for (int j = 0; j < niter; ++j) {
    if (j + 1 < niter)   // issue next V tile into the other buffer (async DMA)
      tdm_load_tile_bf16(vt_lds0 + ((j + 1) & 1) * (32 * DHEAD * 2),
                         Vbase + (size_t)(j + 1) * 32 * QKVC, DHEAD, 32, QKVC);

    // scores: two 16x16 tiles covering keys j*32 .. j*32+31
    v8f s0 = {}, s1 = {};
    const bf16* K0 = qkv + (rowbase + j * 32 + m)      * QKVC + 4096 + kvh * DHEAD;
    const bf16* K1 = qkv + (rowbase + j * 32 + 16 + m) * QKVC + 4096 + kvh * DHEAD;
#pragma unroll
    for (int ks = 0; ks < 4; ++ks) {
      v16bf b0 = frag16(K0 + ks * 32 + hi * 16, K0 + ks * 32 + hi * 16 + 8);
      v16bf b1 = frag16(K1 + ks * 32 + hi * 16, K1 + ks * 32 + hi * 16 + 8);
      s0 = wmma_bf16(qf[ks], b0, s0);
      s1 = wmma_bf16(qf[ks], b1, s1);
    }

    // online softmax (row m = r + 8*hi lives in one 16-lane half)
    float alphav[8];
#pragma unroll
    for (int r = 0; r < 8; ++r) {
      const int qrow = qt * 16 + r + 8 * hi;
      const int k0i = j * 32 + m, k1i = k0i + 16;
      float a0 = (k0i <= qrow) ? s0[r] * ATT_SCALE : NEGINF;
      float a1 = (k1i <= qrow) ? s1[r] * ATT_SCALE : NEGINF;
      float rm = fmaxf(a0, a1);
      rm = fmaxf(rm, __shfl_xor(rm, 1));
      rm = fmaxf(rm, __shfl_xor(rm, 2));
      rm = fmaxf(rm, __shfl_xor(rm, 4));
      rm = fmaxf(rm, __shfl_xor(rm, 8));
      const float mnew = fmaxf(mstat[r], rm);
      const float alpha = __expf(mstat[r] - mnew);
      const float p0 = __expf(a0 - mnew);
      const float p1 = __expf(a1 - mnew);
      float rs = p0 + p1;
      rs += __shfl_xor(rs, 1);
      rs += __shfl_xor(rs, 2);
      rs += __shfl_xor(rs, 4);
      rs += __shfl_xor(rs, 8);
      lstat[r] = lstat[r] * alpha + rs;
      mstat[r] = mnew;
      alphav[r] = alpha;
      pt[(r + 8 * hi) * 32 + m]      = (bf16)p0;   // P in [m][key] layout
      pt[(r + 8 * hi) * 32 + 16 + m] = (bf16)p1;
    }
#pragma unroll
    for (int dt = 0; dt < 8; ++dt)
#pragma unroll
      for (int r = 0; r < 8; ++r) acc[dt][r] *= alphav[r];

    // wait until V tile j is resident (tile j+1 may still be in flight)
    if (j + 1 < niter) {
      __builtin_amdgcn_s_wait_tensorcnt(1);
    } else {
      __builtin_amdgcn_s_wait_tensorcnt(0);
    }
    const bf16* vt = &VT[wave][j & 1][0];

    // P (A-frag from LDS) @ V (B-frags gathered from row-major LDS tile)
    v16bf pf = frag16(pt + m * 32 + hi * 16, pt + m * 32 + hi * 16 + 8);
#pragma unroll
    for (int dt = 0; dt < 8; ++dt) {
      v16bf vf;
#pragma unroll
      for (int e = 0; e < 16; ++e)
        vf[e] = vt[(hi * 16 + e) * DHEAD + dt * 16 + m];
      acc[dt] = wmma_bf16(pf, vf, acc[dt]);
    }
  }

  // finalize and store bf16
#pragma unroll
  for (int dt = 0; dt < 8; ++dt)
#pragma unroll
    for (int r = 0; r < 8; ++r) {
      const size_t row = rowbase + qt * 16 + r + 8 * hi;
      aout[row * (size_t)HID + h * DHEAD + dt * 16 + m] = (bf16)(acc[dt][r] / lstat[r]);
    }
}

// ---- kernel 5: output projection GEMM, 16x64 tile per wave, f32 result -----
// grid = (BS/16, 16), block = 128.

__global__ __launch_bounds__(128) void k_oproj(const bf16* __restrict__ A,   // [BS, HID]
                                               const bf16* __restrict__ WT,  // [HID, HID] (w_o^T)
                                               float* __restrict__ out) {    // [BS, HID]
  const int wave = threadIdx.x >> 5;
  const int lane = threadIdx.x & 31;
  const int rowTile = blockIdx.x;                     // 0..255
  const int colTile0 = (blockIdx.y * 4 + wave) * 4;   // 4 column tiles per wave
  const int m  = lane & 15;
  const int hi = lane >> 4;

  const bf16* Arow = A + (size_t)(rowTile * 16 + m) * HID;
  const bf16* Brow0 = WT + (size_t)((colTile0 + 0) * 16 + m) * HID;
  const bf16* Brow1 = WT + (size_t)((colTile0 + 1) * 16 + m) * HID;
  const bf16* Brow2 = WT + (size_t)((colTile0 + 2) * 16 + m) * HID;
  const bf16* Brow3 = WT + (size_t)((colTile0 + 3) * 16 + m) * HID;

  v8f acc0 = {}, acc1 = {}, acc2 = {}, acc3 = {};
  for (int k0 = 0; k0 < HID; k0 += 32) {
    __builtin_prefetch(Arow + k0 + 512, 0, 0);
    __builtin_prefetch(Brow0 + k0 + 512, 0, 0);
    v16bf a = frag16(Arow + k0 + hi * 16, Arow + k0 + hi * 16 + 8);
    v16bf b0 = frag16(Brow0 + k0 + hi * 16, Brow0 + k0 + hi * 16 + 8);
    v16bf b1 = frag16(Brow1 + k0 + hi * 16, Brow1 + k0 + hi * 16 + 8);
    v16bf b2 = frag16(Brow2 + k0 + hi * 16, Brow2 + k0 + hi * 16 + 8);
    v16bf b3 = frag16(Brow3 + k0 + hi * 16, Brow3 + k0 + hi * 16 + 8);
    acc0 = wmma_bf16(a, b0, acc0);
    acc1 = wmma_bf16(a, b1, acc1);
    acc2 = wmma_bf16(a, b2, acc2);
    acc3 = wmma_bf16(a, b3, acc3);
  }
#pragma unroll
  for (int r = 0; r < 8; ++r) {
    const size_t row = (size_t)(rowTile * 16 + r + 8 * hi);
    out[row * HID + (colTile0 + 0) * 16 + m] = acc0[r];
    out[row * HID + (colTile0 + 1) * 16 + m] = acc1[r];
    out[row * HID + (colTile0 + 2) * 16 + m] = acc2[r];
    out[row * HID + (colTile0 + 3) * 16 + m] = acc3[r];
  }
}

// ---------------------------------------------------------------------------

extern "C" void kernel_launch(void* const* d_in, const int* in_sizes, int n_in,
                              void* d_out, int out_size, void* d_ws, size_t ws_size,
                              hipStream_t stream) {
  const float* hidden = (const float*)d_in[0];   // [B,S,HID]
  const float* w_qkv  = (const float*)d_in[1];   // [HID, QKVC]
  const float* w_o    = (const float*)d_in[2];   // [HID, HID]
  const int*   posns  = (const int*)d_in[3];     // [B,S]
  float* out = (float*)d_out;

  char* ws = (char*)d_ws;
  bf16* Xbf    = (bf16*)(ws);                                  // 33,554,432 B
  bf16* WqkvT  = (bf16*)(ws + (size_t)33554432);               // 50,331,648 B
  bf16* WoT    = (bf16*)(ws + (size_t)83886080);               // 33,554,432 B
  bf16* QKV    = (bf16*)(ws + (size_t)117440512);              // 50,331,648 B
  bf16* AttnO  = (bf16*)(ws + (size_t)167772160);              // 33,554,432 B

  k_cvt_bf16<<<dim3((BS * HID) / 4 / 256), dim3(256), 0, stream>>>(hidden, Xbf, BS * HID);
  k_transpose_cvt<<<dim3(QKVC / 32, HID / 32), dim3(256), 0, stream>>>(w_qkv, WqkvT, HID, QKVC);
  k_transpose_cvt<<<dim3(HID / 32, HID / 32), dim3(256), 0, stream>>>(w_o, WoT, HID, HID);
  k_qkv_rope<<<dim3(BS / 16, 48), dim3(128), 0, stream>>>(Xbf, WqkvT, posns, QKV);
  k_attn<<<dim3(SEQ / 16 / 4, 2 * NHEAD), dim3(128), 0, stream>>>(QKV, AttnO);
  k_oproj<<<dim3(BS / 16, 16), dim3(128), 0, stream>>>(AttnO, WoT, out);
}
